// TPMoELayer_15427522527441
// MI455X (gfx1250) — matrix operand order repaired and checked by reading, not compiled, not run on hardware
//
#include <hip/hip_runtime.h>

// ---------------------------------------------------------------------------
// MoE top-2 SwiGLU for MI455X (gfx1250): gather-compute + bf16 WMMA,
// B-tiles staged in LDS via async global->LDS copies (double buffered).
// ---------------------------------------------------------------------------

typedef __attribute__((ext_vector_type(16))) __bf16        v16bf;
typedef __attribute__((ext_vector_type(8)))  float         v8f;
typedef __attribute__((ext_vector_type(4)))  unsigned int  u32x4;
typedef __attribute__((ext_vector_type(2)))  unsigned int  u32x2;
typedef __attribute__((ext_vector_type(4)))  float         f32x4;
typedef __attribute__((vector_size(16)))     int           v4i_t;   // builtin param type

#define DEVI __device__ __forceinline__
#define AS1 __attribute__((address_space(1)))
#define AS3 __attribute__((address_space(3)))

constexpr int T_TOK = 8192;   // B*S
constexpr int H_DIM = 1024;
constexpr int I_DIM = 2816;
constexpr int E_NUM = 8;

constexpr int KC    = 64;     // K-chunk staged per LDS buffer
constexpr int PADB  = 144;    // bytes per column in LDS (128 data + 16 pad)
constexpr int BMATB = 128 * PADB;   // one staged B matrix chunk: 18432 B

// Workspace layout (bytes), all 256-aligned by construction.
constexpr size_t OFF_XB = 0;                                          // bf16 x      [T][H]
constexpr size_t OFF_WG = OFF_XB + (size_t)T_TOK * H_DIM * 2;         // bf16 w_gate [E][I][H]
constexpr size_t OFF_WU = OFF_WG + (size_t)E_NUM * I_DIM * H_DIM * 2;
constexpr size_t OFF_WD = OFF_WU + (size_t)E_NUM * I_DIM * H_DIM * 2; // bf16 w_down [E][H][I]
constexpr size_t OFF_HB = OFF_WD + (size_t)E_NUM * I_DIM * H_DIM * 2; // bf16 hidden [2T][I]
constexpr size_t OFF_DW = OFF_HB + (size_t)2 * T_TOK * I_DIM * 2;     // f32 down    [2T][H]
constexpr size_t OFF_WT = OFF_DW + (size_t)2 * T_TOK * H_DIM * 4;     // f32 weight  [2T]
constexpr size_t OFF_LS = OFF_WT + (size_t)2 * T_TOK * 4;             // int list    [E][T]
constexpr size_t OFF_CT = OFF_LS + (size_t)E_NUM * T_TOK * 4;         // int cnt     [E]

union FragBF { v16bf v; u32x4 q[2]; };

DEVI unsigned short f2bf(float f) {
  union { float f; unsigned u; } v; v.f = f;
  unsigned r = v.u + 0x7FFFu + ((v.u >> 16) & 1u);   // round-to-nearest-even
  return (unsigned short)(r >> 16);
}
DEVI unsigned pack2bf(float a, float b) {
  return (unsigned)f2bf(a) | ((unsigned)f2bf(b) << 16);
}

DEVI v8f wmma_bf16(v16bf a, v16bf b, v8f c) {
  // D(16x16,f32) = A(16x32,bf16) x B(32x16,bf16) + C
  return __builtin_amdgcn_wmma_f32_16x16x32_bf16(false, a, false, b, (short)0, c,
                                                 false, false);
}

// --------------------------- async copy plumbing ---------------------------
#if __has_builtin(__builtin_amdgcn_global_load_async_to_lds_b128)
#define HAS_ASYNC 1
#else
#define HAS_ASYNC 0
#endif

#if HAS_ASYNC
#if __has_builtin(__builtin_amdgcn_s_wait_asynccnt)
#define WAIT_ASYNC(n) __builtin_amdgcn_s_wait_asynccnt((unsigned short)(n))
#else
#define WAIT_ASYNC(n) asm volatile("s_wait_asynccnt %0" ::"n"(n) : "memory")
#endif
#else
#define WAIT_ASYNC(n) ((void)0)
#endif

// Copy 16 bytes global -> LDS. Async engine path when available (ASYNCcnt),
// else a plain synchronous reg->ds copy (correct fallback).
DEVI void cp16(const unsigned short* g, unsigned char* l) {
#if HAS_ASYNC
  __builtin_amdgcn_global_load_async_to_lds_b128(
      (AS1 v4i_t*)g, (AS3 v4i_t*)l, 0, 0);
#else
  *(u32x4*)l = *(const u32x4*)g;
#endif
}

// ---------------------------------------------------------------------------
// fp32 -> bf16 bulk convert (16B in / 8B out per thread)
// ---------------------------------------------------------------------------
__global__ __launch_bounds__(256) void cvt_bf16_kernel(const f32x4* __restrict__ in,
                                                       u32x2* __restrict__ out, int n4) {
  int i = blockIdx.x * 256 + threadIdx.x;
  if (i >= n4) return;
  f32x4 v = in[i];
  u32x2 o;
  o[0] = pack2bf(v[0], v[1]);
  o[1] = pack2bf(v[2], v[3]);
  out[i] = o;
}

// ---------------------------------------------------------------------------
// Router: one wave (32 lanes) per token. w_router staged in LDS (32 KB).
// Produces renormalized top-2 weights (per slot = 2t+k) and expert token lists.
// ---------------------------------------------------------------------------
__global__ __launch_bounds__(256) void router_kernel(const float* __restrict__ x,
                                                     const float* __restrict__ wr,
                                                     float* __restrict__ weight,
                                                     int* __restrict__ list,
                                                     int* __restrict__ cnt) {
  __shared__ float wl[E_NUM * H_DIM];
  for (int i = threadIdx.x; i < E_NUM * H_DIM; i += 256) wl[i] = wr[i];
  __syncthreads();

  const int wave = threadIdx.x >> 5;
  const int lane = threadIdx.x & 31;
  const int t = blockIdx.x * 8 + wave;
  const float* xr = x + (size_t)t * H_DIM;

  float acc[E_NUM];
#pragma unroll
  for (int e = 0; e < E_NUM; ++e) acc[e] = 0.f;

  for (int h = lane; h < H_DIM; h += 32) {
    float xv = xr[h];
#pragma unroll
    for (int e = 0; e < E_NUM; ++e) acc[e] = fmaf(xv, wl[e * H_DIM + h], acc[e]);
  }
#pragma unroll
  for (int e = 0; e < E_NUM; ++e) {
#pragma unroll
    for (int off = 16; off > 0; off >>= 1) acc[e] += __shfl_xor(acc[e], off, 32);
  }

  if (lane == 0) {
    float mx = acc[0];
#pragma unroll
    for (int e = 1; e < E_NUM; ++e) mx = fmaxf(mx, acc[e]);
    float p[E_NUM];
#pragma unroll
    for (int e = 0; e < E_NUM; ++e) p[e] = __expf(acc[e] - mx);
    int e0 = 0;
#pragma unroll
    for (int e = 1; e < E_NUM; ++e) if (p[e] > p[e0]) e0 = e;
    int e1 = (e0 == 0) ? 1 : 0;
#pragma unroll
    for (int e = 0; e < E_NUM; ++e) if (e != e0 && p[e] > p[e1]) e1 = e;
    float inv = 1.f / (p[e0] + p[e1]);   // softmax denom cancels in renorm
    int pos0 = atomicAdd(&cnt[e0], 1);
    int pos1 = atomicAdd(&cnt[e1], 1);
    list[e0 * T_TOK + pos0] = 2 * t;
    list[e1 * T_TOK + pos1] = 2 * t + 1;
    weight[2 * t]     = p[e0] * inv;
    weight[2 * t + 1] = p[e1] * inv;
  }
}

// ---------------------------------------------------------------------------
// Fragment loads (ISA 7.12.2 bf16 layouts, wave32).
// A (16x32, MxK): lane = m, lane-half selects interleaved K octets.
// B (32x16, KxN): lane = n, lane-half selects interleaved K octets.
// ---------------------------------------------------------------------------
DEVI FragBF load_a(const unsigned short* p, int k) {   // p = row base + 8*half
  FragBF f;
  f.q[0] = *(const u32x4*)(p + k);
  f.q[1] = *(const u32x4*)(p + k + 16);
  return f;
}
DEVI FragBF lds_b_frag(const unsigned char* p) {       // p = staged col + 64*s + 16*half
  FragBF f;
  f.q[0] = *(const u32x4*)p;
  f.q[1] = *(const u32x4*)(p + 32);
  return f;
}

// ---------------------------------------------------------------------------
// GEMM1: per expert, gathered tokens. gate/up fused (shared A-frags).
// Block tile 128(M) x 128(N of I), K = H. 8 waves of 32x64.
// B tiles (gate+up) async-staged to LDS, double buffered, KC=64.
// Epilogue: hb[slot][i] = bf16( silu(gate) * up ).
// ---------------------------------------------------------------------------
__global__ __launch_bounds__(256) void gemm1_kernel(const unsigned short* __restrict__ xb,
                                                    const unsigned short* __restrict__ wg,
                                                    const unsigned short* __restrict__ wu,
                                                    const int* __restrict__ list,
                                                    const int* __restrict__ cnt,
                                                    unsigned short* __restrict__ hb) {
  const int e = blockIdx.z;
  const int count = cnt[e];
  const int m_base = blockIdx.x * 128;
  if (m_base >= count) return;
  const int n_base = blockIdx.y * 128;

  const int* lst = list + e * T_TOK;
  const unsigned short* wge = wg + (size_t)e * I_DIM * H_DIM;
  const unsigned short* wue = wu + (size_t)e * I_DIM * H_DIM;

  __shared__ __align__(16) unsigned char lds_b[2][2][BMATB];   // [buf][gate/up] 72 KB

  const int tid = threadIdx.x;
  const int wid = tid >> 5, lane = tid & 31;
  const int half = lane >> 4, r = lane & 15;
  const int mw = (wid & 3) * 32;   // wave's M offset in block tile
  const int nw = (wid >> 2) * 64;  // wave's N offset in block tile

  const unsigned short* aptr[2];
#pragma unroll
  for (int mi = 0; mi < 2; ++mi) {
    int row = m_base + mw + mi * 16 + r;
    row = row < count ? row : count - 1;
    int tok = lst[row] >> 1;
    aptr[mi] = xb + (size_t)tok * H_DIM + 8 * half;
  }

  // Stage one KC-chunk of both B matrices (gate, up): 2*128 cols * 128B.
  // 2048 16B pieces / 256 threads = 8 async copies per thread per chunk.
  auto stage = [&](int bufi, int kb) {
    unsigned char* base = &lds_b[bufi][0][0];
#pragma unroll
    for (int it = 0; it < 8; ++it) {
      int p = tid + it * 256;
      int mat = p >> 10;
      int rem = p & 1023;
      int col = rem >> 3;
      int seg = rem & 7;
      const unsigned short* g =
          (mat ? wue : wge) + (size_t)(n_base + col) * H_DIM + kb + seg * 8;
      cp16(g, base + mat * BMATB + col * PADB + seg * 16);
    }
  };

  v8f cg[2][4] = {};
  v8f cu[2][4] = {};

  constexpr int NCH = H_DIM / KC;   // 16
  stage(0, 0);
  for (int cc = 0; cc < NCH; ++cc) {
    const int buf = cc & 1;
    if (cc + 1 < NCH) {
      stage(buf ^ 1, (cc + 1) * KC);
      WAIT_ASYNC(8);                 // drain chunk cc, keep cc+1 in flight
    } else {
      WAIT_ASYNC(0);
    }
    __syncthreads();

    const unsigned char* bG = &lds_b[buf][0][0];
    const unsigned char* bU = &lds_b[buf][1][0];
#pragma unroll
    for (int s = 0; s < 2; ++s) {
      const int k = cc * KC + s * 32;
      FragBF a0 = load_a(aptr[0], k);
      FragBF a1 = load_a(aptr[1], k);
#pragma unroll
      for (int ni = 0; ni < 4; ++ni) {
        const int loff = (nw + ni * 16 + r) * PADB + s * 64 + half * 16;
        FragBF g = lds_b_frag(bG + loff);
        FragBF u = lds_b_frag(bU + loff);
        cg[0][ni] = wmma_bf16(a0.v, g.v, cg[0][ni]);
        cg[1][ni] = wmma_bf16(a1.v, g.v, cg[1][ni]);
        cu[0][ni] = wmma_bf16(a0.v, u.v, cu[0][ni]);
        cu[1][ni] = wmma_bf16(a1.v, u.v, cu[1][ni]);
      }
    }
    __syncthreads();                 // readers done before buf is re-staged
  }

  // C layout: lane r = column; VGPR j holds row (8*half + j).
#pragma unroll
  for (int mi = 0; mi < 2; ++mi) {
#pragma unroll
    for (int j = 0; j < 8; ++j) {
      int row = m_base + mw + mi * 16 + 8 * half + j;
      if (row < count) {
        int slot = lst[row];
        unsigned short* hrow = hb + (size_t)slot * I_DIM;
#pragma unroll
        for (int ni = 0; ni < 4; ++ni) {
          int col = n_base + nw + ni * 16 + r;
          float g = cg[mi][ni][j];
          float u = cu[mi][ni][j];
          float hv = (g / (1.f + __expf(-g))) * u;   // silu(g) * u
          hrow[col] = f2bf(hv);
        }
      }
    }
  }
}

// ---------------------------------------------------------------------------
// GEMM2: down projection. A = hb rows (K = I), B = w_down[e][h][i] staged in LDS.
// Epilogue scales by renormalized router weight, writes per-slot f32 rows.
// ---------------------------------------------------------------------------
__global__ __launch_bounds__(256) void gemm2_kernel(const unsigned short* __restrict__ hb,
                                                    const unsigned short* __restrict__ wd,
                                                    const int* __restrict__ list,
                                                    const int* __restrict__ cnt,
                                                    const float* __restrict__ weight,
                                                    float* __restrict__ dws) {
  const int e = blockIdx.z;
  const int count = cnt[e];
  const int m_base = blockIdx.x * 128;
  if (m_base >= count) return;
  const int n_base = blockIdx.y * 128;   // over H

  const int* lst = list + e * T_TOK;
  const unsigned short* wde = wd + (size_t)e * H_DIM * I_DIM;

  __shared__ __align__(16) unsigned char lds_b[2][BMATB];      // 36 KB

  const int tid = threadIdx.x;
  const int wid = tid >> 5, lane = tid & 31;
  const int half = lane >> 4, r = lane & 15;
  const int mw = (wid & 3) * 32;
  const int nw = (wid >> 2) * 64;

  const unsigned short* aptr[2];
#pragma unroll
  for (int mi = 0; mi < 2; ++mi) {
    int row = m_base + mw + mi * 16 + r;
    row = row < count ? row : count - 1;
    int slot = lst[row];
    aptr[mi] = hb + (size_t)slot * I_DIM + 8 * half;
  }

  // 1024 16B pieces / 256 threads = 4 async copies per thread per chunk.
  auto stage = [&](int bufi, int kb) {
    unsigned char* base = &lds_b[bufi][0];
#pragma unroll
    for (int it = 0; it < 4; ++it) {
      int p = tid + it * 256;
      int col = p >> 3;
      int seg = p & 7;
      const unsigned short* g = wde + (size_t)(n_base + col) * I_DIM + kb + seg * 8;
      cp16(g, base + col * PADB + seg * 16);
    }
  };

  v8f cc_[2][4] = {};

  constexpr int NCH = I_DIM / KC;   // 44
  stage(0, 0);
  for (int cc = 0; cc < NCH; ++cc) {
    const int buf = cc & 1;
    if (cc + 1 < NCH) {
      stage(buf ^ 1, (cc + 1) * KC);
      WAIT_ASYNC(4);
    } else {
      WAIT_ASYNC(0);
    }
    __syncthreads();

    const unsigned char* bB = &lds_b[buf][0];
#pragma unroll
    for (int s = 0; s < 2; ++s) {
      const int k = cc * KC + s * 32;
      FragBF a0 = load_a(aptr[0], k);
      FragBF a1 = load_a(aptr[1], k);
#pragma unroll
      for (int ni = 0; ni < 4; ++ni) {
        const int loff = (nw + ni * 16 + r) * PADB + s * 64 + half * 16;
        FragBF b = lds_b_frag(bB + loff);
        cc_[0][ni] = wmma_bf16(a0.v, b.v, cc_[0][ni]);
        cc_[1][ni] = wmma_bf16(a1.v, b.v, cc_[1][ni]);
      }
    }
    __syncthreads();
  }

#pragma unroll
  for (int mi = 0; mi < 2; ++mi) {
#pragma unroll
    for (int j = 0; j < 8; ++j) {
      int row = m_base + mw + mi * 16 + 8 * half + j;
      if (row < count) {
        int slot = lst[row];
        float wgt = weight[slot];
        float* drow = dws + (size_t)slot * H_DIM;
#pragma unroll
        for (int ni = 0; ni < 4; ++ni) {
          int col = n_base + nw + ni * 16 + r;
          drow[col] = cc_[mi][ni][j] * wgt;
        }
      }
    }
  }
}

// ---------------------------------------------------------------------------
// Combine: out[t][h] = dws[2t][h] + dws[2t+1][h]   (fully overwrites d_out)
// ---------------------------------------------------------------------------
__global__ __launch_bounds__(256) void combine_kernel(const float* __restrict__ dws,
                                                      float* __restrict__ out) {
  int i = blockIdx.x * 256 + threadIdx.x;         // over T*H/4
  int t = i / (H_DIM / 4);
  int h4 = i - t * (H_DIM / 4);
  const f32x4* a = (const f32x4*)(dws + (size_t)(2 * t) * H_DIM) + h4;
  const f32x4* b = (const f32x4*)(dws + (size_t)(2 * t + 1) * H_DIM) + h4;
  ((f32x4*)out)[i] = *a + *b;
}

// ---------------------------------------------------------------------------
extern "C" void kernel_launch(void* const* d_in, const int* in_sizes, int n_in,
                              void* d_out, int out_size, void* d_ws, size_t ws_size,
                              hipStream_t stream) {
  const float* x  = (const float*)d_in[0];   // [4,2048,1024]
  const float* wr = (const float*)d_in[1];   // [8,1024]
  const float* wg = (const float*)d_in[2];   // [8,2816,1024]
  const float* wu = (const float*)d_in[3];   // [8,2816,1024]
  const float* wd = (const float*)d_in[4];   // [8,1024,2816]
  float* out = (float*)d_out;

  char* ws = (char*)d_ws;
  unsigned short* xb  = (unsigned short*)(ws + OFF_XB);
  unsigned short* wgb = (unsigned short*)(ws + OFF_WG);
  unsigned short* wub = (unsigned short*)(ws + OFF_WU);
  unsigned short* wdb = (unsigned short*)(ws + OFF_WD);
  unsigned short* hb  = (unsigned short*)(ws + OFF_HB);
  float* dws    = (float*)(ws + OFF_DW);
  float* weight = (float*)(ws + OFF_WT);
  int*   list   = (int*)(ws + OFF_LS);
  int*   cnt    = (int*)(ws + OFF_CT);

  (void)hipMemsetAsync(cnt, 0, E_NUM * sizeof(int), stream);

  // fp32 -> bf16 conversions
  {
    int n4 = T_TOK * H_DIM / 4;
    cvt_bf16_kernel<<<(n4 + 255) / 256, 256, 0, stream>>>((const f32x4*)x, (u32x2*)xb, n4);
  }
  {
    int n4 = E_NUM * I_DIM * H_DIM / 4;
    cvt_bf16_kernel<<<(n4 + 255) / 256, 256, 0, stream>>>((const f32x4*)wg, (u32x2*)wgb, n4);
    cvt_bf16_kernel<<<(n4 + 255) / 256, 256, 0, stream>>>((const f32x4*)wu, (u32x2*)wub, n4);
    cvt_bf16_kernel<<<(n4 + 255) / 256, 256, 0, stream>>>((const f32x4*)wd, (u32x2*)wdb, n4);
  }

  // Router (one wave per token)
  router_kernel<<<T_TOK / 8, 256, 0, stream>>>(x, wr, weight, list, cnt);

  // Expert GEMMs
  dim3 g1(T_TOK / 128, I_DIM / 128, E_NUM);   // (64, 22, 8)
  gemm1_kernel<<<g1, 256, 0, stream>>>(xb, wgb, wub, list, cnt, hb);

  dim3 g2(T_TOK / 128, H_DIM / 128, E_NUM);   // (64, 8, 8)
  gemm2_kernel<<<g2, 256, 0, stream>>>(hb, wdb, list, cnt, weight, dws);

  // Combine the two slots per token
  combine_kernel<<<(T_TOK * H_DIM / 4 + 255) / 256, 256, 0, stream>>>(dws, out);
}
